// KNNRetrieval_53094385713848
// MI455X (gfx1250) — matrix operand order repaired and checked by reading, not compiled, not run on hardware
//
#include <hip/hip_runtime.h>

typedef _Float16 v8h  __attribute__((ext_vector_type(8)));
typedef _Float16 v16h __attribute__((ext_vector_type(16)));
typedef float    v8f  __attribute__((ext_vector_type(8)));

#define NQ     2048
#define NS     65536
#define F      512
#define KNN    16
#define SEG    8
#define NS_SEG (NS / SEG)     // 8192
#define MB     32             // queries per workgroup
#define NB     64             // support rows per chunk
#define ITERS  (NS_SEG / NB)  // 128
#define KB_N   (F / 32)       // 16 K-blocks per tile

// LDS layout (bytes). Row stride 1040 = 1024 + 16B pad -> conflict-free b128.
#define QSTR      1040
#define Q_OFF     0
#define S0_OFF    (MB * QSTR)                 // 33280
#define SBUF_SZ   (NB * QSTR)                 // 66560 (double-buffered support)
#define DUMP_OFF  (S0_OFF + 2 * SBUF_SZ)      // 166400
#define DUMP_WSTR 1280                        // 16 rows * 80B per wave
#define DUMP_RSTR 80
#define LDS_BYTES (DUMP_OFF + 8 * DUMP_WSTR)  // 176640 (~172 KB, 1 WG / WGP)
#define MERGE_OFF S0_OFF                      // reuse support buf0 after loop

static __device__ inline unsigned pk2h(float a, float b) {
    _Float16 ha = (_Float16)a, hb = (_Float16)b;
    unsigned short ua = __builtin_bit_cast(unsigned short, ha);
    unsigned short ub = __builtin_bit_cast(unsigned short, hb);
    return (unsigned)ua | ((unsigned)ub << 16);
}

static __device__ inline v16h make_v16h(uint4 a, uint4 b) {
    v8h lo = __builtin_bit_cast(v8h, a);
    v8h hi = __builtin_bit_cast(v8h, b);
    return __builtin_shufflevector(lo, hi, 0,1,2,3,4,5,6,7,8,9,10,11,12,13,14,15);
}

// Issue one support chunk (NB rows x 1024B) as 16 async b128 per thread:
// global (saddr base + 32b voffset) -> LDS (32b byte-offset VGPR). ASYNCcnt.
static __device__ inline void issue_chunk_async(
    const unsigned short* __restrict__ sh, unsigned ldsBase,
    int srow0, int bufOff, int tid)
{
    #pragma unroll
    for (int i = 0; i < 16; ++i) {
        int c   = tid + i * 256;            // 4096 16B chunks
        int r   = c >> 6;                   // 64 chunks per row
        int off = (c & 63) * 16;            // byte offset in row
        unsigned lds = ldsBase + (unsigned)(bufOff + r * QSTR + off);
        unsigned gof = (unsigned)((srow0 + r) * (F * 2) + off);
        asm volatile("global_load_async_to_lds_b128 %0, %1, %2"
                     :: "v"(lds), "v"(gof), "s"(sh) : "memory");
    }
}

// ---------------------------------------------------------------------------
// Kernel 0: f32 rows -> packed f16 rows + exact f32 squared norms.
// One wave32 per row; lane handles 16 contiguous elements.
// ---------------------------------------------------------------------------
__global__ __launch_bounds__(256) void knn_convert_rows(
    const float* __restrict__ src, unsigned short* __restrict__ dst,
    float* __restrict__ norms, int nrows)
{
    const int wid  = (int)((blockIdx.x * blockDim.x + threadIdx.x) >> 5);
    const int lane = (int)(threadIdx.x & 31);
    if (wid >= nrows) return;

    const float4* p = (const float4*)(src + (size_t)wid * F + lane * 16);
    float4 a = p[0], b = p[1], c = p[2], d = p[3];
    float ss = a.x*a.x + a.y*a.y + a.z*a.z + a.w*a.w
             + b.x*b.x + b.y*b.y + b.z*b.z + b.w*b.w
             + c.x*c.x + c.y*c.y + c.z*c.z + c.w*c.w
             + d.x*d.x + d.y*d.y + d.z*d.z + d.w*d.w;

    uint4* o = (uint4*)(dst + (size_t)wid * F + lane * 16);
    o[0] = make_uint4(pk2h(a.x,a.y), pk2h(a.z,a.w), pk2h(b.x,b.y), pk2h(b.z,b.w));
    o[1] = make_uint4(pk2h(c.x,c.y), pk2h(c.z,c.w), pk2h(d.x,d.y), pk2h(d.z,d.w));

    #pragma unroll
    for (int off = 16; off > 0; off >>= 1) ss += __shfl_xor(ss, off, 32);
    if (lane == 0) norms[wid] = ss;
}

// ---------------------------------------------------------------------------
// Kernel 1: WMMA f16 GEMM tiles + streaming per-segment top-16, with
// double-buffered GLOBAL_LOAD_ASYNC_TO_LDS_B128 support staging and
// software-pipelined fragment loads.
// grid = (NQ/MB, SEG); 256 threads = 8 waves = 2 M-tiles x 4 N-tiles.
// ---------------------------------------------------------------------------
__global__ __launch_bounds__(256) void knn_gemm_topk(
    const unsigned short* __restrict__ qh, const unsigned short* __restrict__ sh,
    const float* __restrict__ q2, const float* __restrict__ s2,
    uint2* __restrict__ parts)
{
    extern __shared__ char smem[];
    const unsigned ldsBase = (unsigned)(size_t)(void*)smem;  // low 32 = LDS offset
    const int tid  = (int)threadIdx.x;
    const int lane = tid & 31;
    const int w    = tid >> 5;
    const int mt   = w >> 2;   // 0..1
    const int nt   = w & 3;    // 0..3
    const int qb   = (int)blockIdx.x;
    const int seg  = (int)blockIdx.y;
    const int segbase = seg * NS_SEG;
    const int half = lane >> 4;
    const int nl   = lane & 15;

    // Preload resident query tile (MB x F halves).
    {
        const size_t qrow0 = (size_t)qb * MB;
        #pragma unroll
        for (int i = 0; i < 8; ++i) {
            int c   = tid + i * 256;       // 2048 16B chunks
            int r   = c >> 6;              // 64 chunks per row
            int off = (c & 63) * 16;       // byte offset in row
            uint4 v = *(const uint4*)((const char*)(qh + (qrow0 + r) * F) + off);
            *(uint4*)(smem + Q_OFF + r * QSTR + off) = v;
        }
    }

    // Loop-invariant query norms for this lane's 8 C rows (M = r + 8*half).
    float qnr[8];
    #pragma unroll
    for (int r = 0; r < 8; ++r)
        qnr[r] = q2[qb * MB + mt * 16 + r + 8 * half];

    // Prologue of the async pipeline: chunk 0 -> buffer 0.
    issue_chunk_async(sh, ldsBase, segbase, S0_OFF, tid);

    // Per-lane sorted top-16 (register resident), for query row lane&15 of the
    // wave's M-tile, scanning columns (lane>>4)*8 .. +7 of the wave's N-tile.
    float bd[KNN]; int bi[KNN];
    #pragma unroll
    for (int e = 0; e < KNN; ++e) { bd[e] = 3.0e38f; bi[e] = 0; }

    for (int it = 0; it < ITERS; ++it) {
        const int cur   = it & 1;
        const int srow0 = segbase + it * NB;

        // Issue next chunk into the other buffer (safe: all waves finished
        // reading it before the previous iteration's dump barrier).
        if (it + 1 < ITERS)
            issue_chunk_async(sh, ldsBase, srow0 + NB,
                              S0_OFF + (cur ^ 1) * SBUF_SZ, tid);
        // Support norm for this lane's column (L2-resident; latency hides
        // under the WMMA loop).
        const float s2v = s2[srow0 + nt * 16 + nl];
        if (it + 2 < ITERS)   // warm L2 for the chunk after next
            __builtin_prefetch(sh + (size_t)(srow0 + 2 * NB) * F + tid * 128, 0, 0);

        // Each thread issued exactly 16 async b128 per chunk; async loads
        // complete in order, so <=16 outstanding => current chunk landed.
        if (it + 1 < ITERS) asm volatile("s_wait_asynccnt 0x10" ::: "memory");
        else                asm volatile("s_wait_asynccnt 0x0"  ::: "memory");
        __syncthreads();

        // --- WMMA accumulation over K = 512 (16 x 16x16x32_f16),
        //     fragments software-pipelined so ds_load latency overlaps WMMA ---
        v8f acc = {};
        const char* aBase = smem + Q_OFF + (mt * 16 + nl) * QSTR + half * 32;
        const char* bBase = smem + S0_OFF + cur * SBUF_SZ
                          + (nt * 16 + nl) * QSTR + half * 32;
        uint4 ca0 = *(const uint4*)(aBase);
        uint4 ca1 = *(const uint4*)(aBase + 16);
        uint4 cb0 = *(const uint4*)(bBase);
        uint4 cb1 = *(const uint4*)(bBase + 16);
        #pragma unroll
        for (int kb = 0; kb < KB_N; ++kb) {
            uint4 na0 = ca0, na1 = ca1, nb0 = cb0, nb1 = cb1;
            if (kb + 1 < KB_N) {
                const int o = (kb + 1) * 64;              // k0 * 2 bytes
                na0 = *(const uint4*)(aBase + o);         // K k0+16h  .. +7
                na1 = *(const uint4*)(aBase + o + 16);    // K k0+16h+8.. +7
                nb0 = *(const uint4*)(bBase + o);         // K k0+16h  .. +7
                nb1 = *(const uint4*)(bBase + o + 16);    // K k0+16h+8.. +7
            }
            acc = __builtin_amdgcn_wmma_f32_16x16x32_f16(
                false, make_v16h(ca0, ca1), false, make_v16h(cb0, cb1),
                (short)0, acc, false, false);
            ca0 = na0; ca1 = na1; cb0 = nb0; cb1 = nb1;
        }

        // --- d2 = |q|^2 + |s|^2 - 2 dot ; dump 16x16 tile to per-wave LDS ---
        {
            char* dump = smem + DUMP_OFF + w * DUMP_WSTR;
            #pragma unroll
            for (int r = 0; r < 8; ++r) {
                int mrow = r + 8 * half;   // C layout: VGPR r -> M=r / M=r+8
                float d2v = qnr[r] + s2v - 2.0f * acc[r];
                *(float*)(dump + mrow * DUMP_RSTR + nl * 4) = d2v;
            }
        }
        __syncthreads();  // cross-lane visibility of the dump

        // --- streaming top-16 insertion (threshold test prunes ~all) ---
        {
            const char* dump = smem + DUMP_OFF + w * DUMP_WSTR;
            const int row = lane & 15;
            const int cb  = half * 8;
            const uint4* dr = (const uint4*)(dump + row * DUMP_RSTR + cb * 4);
            uint4 u0 = dr[0], u1 = dr[1];       // two ds_load_b128
            float cand[8] = {
                __builtin_bit_cast(float, u0.x), __builtin_bit_cast(float, u0.y),
                __builtin_bit_cast(float, u0.z), __builtin_bit_cast(float, u0.w),
                __builtin_bit_cast(float, u1.x), __builtin_bit_cast(float, u1.y),
                __builtin_bit_cast(float, u1.z), __builtin_bit_cast(float, u1.w) };
            const int idx0 = srow0 + nt * 16 + cb;
            #pragma unroll
            for (int j = 0; j < 8; ++j) {
                float d = cand[j];
                if (d < bd[KNN - 1]) {
                    float cd = d; int ci = idx0 + j;
                    #pragma unroll
                    for (int t = 0; t < KNN; ++t) {
                        if (cd < bd[t]) {
                            float td = bd[t]; bd[t] = cd; cd = td;
                            int   ti = bi[t]; bi[t] = ci; ci = ti;
                        }
                    }
                }
            }
        }
    }

    // --- workgroup merge: 8 lists of 16 per query -> per-segment top-16 ---
    __syncthreads();
    {
        const int qloc = mt * 16 + nl;
        const int list = nt * 2 + half;
        uint2* m = (uint2*)(smem + MERGE_OFF);
        #pragma unroll
        for (int e = 0; e < KNN; ++e)
            m[(qloc * 8 + list) * KNN + e] =
                make_uint2(__builtin_bit_cast(unsigned, bd[e]), (unsigned)bi[e]);
    }
    __syncthreads();

    if (tid < MB) {
        const uint2* m = (const uint2*)(smem + MERGE_OFF);
        float fd[KNN]; int fi[KNN];
        #pragma unroll
        for (int e = 0; e < KNN; ++e) { fd[e] = 3.0e38f; fi[e] = 0; }
        for (int c = 0; c < 8 * KNN; ++c) {
            uint2 ent = m[tid * 8 * KNN + c];
            float d = __builtin_bit_cast(float, ent.x);
            if (d < fd[KNN - 1]) {
                float cd = d; int ci = (int)ent.y;
                #pragma unroll
                for (int t = 0; t < KNN; ++t) {
                    if (cd < fd[t]) {
                        float td = fd[t]; fd[t] = cd; cd = td;
                        int   ti = fi[t]; fi[t] = ci; ci = ti;
                    }
                }
            }
        }
        uint2* out = parts + ((size_t)seg * NQ + (size_t)qb * MB + tid) * KNN;
        #pragma unroll
        for (int e = 0; e < KNN; ++e)
            out[e] = make_uint2(__builtin_bit_cast(unsigned, fd[e]), (unsigned)fi[e]);
    }
}

// ---------------------------------------------------------------------------
// Kernel 2: merge 8 segment partials per query; emit indices + weights.
// ---------------------------------------------------------------------------
__global__ __launch_bounds__(256) void knn_finalize(
    const uint2* __restrict__ parts, int* __restrict__ idx_out,
    float* __restrict__ w_out)
{
    const int q = (int)(blockIdx.x * blockDim.x + threadIdx.x);
    if (q >= NQ) return;

    float fd[KNN]; int fi[KNN];
    #pragma unroll
    for (int e = 0; e < KNN; ++e) { fd[e] = 3.0e38f; fi[e] = 0; }

    for (int s = 0; s < SEG; ++s) {
        const uint2* p = parts + ((size_t)s * NQ + q) * KNN;
        #pragma unroll
        for (int e = 0; e < KNN; ++e) {
            uint2 ent = p[e];
            float d = __builtin_bit_cast(float, ent.x);
            if (d < fd[KNN - 1]) {
                float cd = d; int ci = (int)ent.y;
                #pragma unroll
                for (int t = 0; t < KNN; ++t) {
                    if (cd < fd[t]) {
                        float td = fd[t]; fd[t] = cd; cd = td;
                        int   ti = fi[t]; fi[t] = ci; ci = ti;
                    }
                }
            }
        }
    }

    float sim[KNN]; float tot = 0.0f;
    #pragma unroll
    for (int e = 0; e < KNN; ++e) {
        float dist = sqrtf(fmaxf(fd[e], 1e-12f));
        sim[e] = 1.0f / (dist + 1e-6f);
        tot += sim[e];
    }
    const float inv = 1.0f / tot;
    #pragma unroll
    for (int e = 0; e < KNN; ++e) {
        idx_out[q * KNN + e] = fi[e];
        w_out[q * KNN + e]   = sim[e] * inv;
    }
}

// ---------------------------------------------------------------------------
// Workspace layout (bytes):
//   [0)     : support f16   NS*F*2  = 64 MB
//   [+64Mi) : query   f16   NQ*F*2  =  2 MB
//   [+66Mi) : support norms NS*4, query norms NQ*4
//   [+..]   : partials SEG*NQ*KNN*8 = 2 MB      (total ~68.3 MB)
// ---------------------------------------------------------------------------
extern "C" void kernel_launch(void* const* d_in, const int* in_sizes, int n_in,
                              void* d_out, int out_size, void* d_ws, size_t ws_size,
                              hipStream_t stream) {
    (void)in_sizes; (void)n_in; (void)out_size; (void)ws_size;
    const float* q = (const float*)d_in[0];
    const float* s = (const float*)d_in[1];

    char* ws = (char*)d_ws;
    unsigned short* sf16 = (unsigned short*)ws;
    unsigned short* qf16 = (unsigned short*)(ws + (size_t)NS * F * 2);
    float* s2 = (float*)(ws + (size_t)NS * F * 2 + (size_t)NQ * F * 2);
    float* q2 = s2 + NS;
    uint2* parts = (uint2*)(q2 + NQ);

    int* idx_out = (int*)d_out;
    float* w_out = (float*)d_out + (size_t)NQ * KNN;

    knn_convert_rows<<<NS / 8, 256, 0, stream>>>(s, sf16, s2, NS);
    knn_convert_rows<<<NQ / 8, 256, 0, stream>>>(q, qf16, q2, NQ);
    knn_gemm_topk<<<dim3(NQ / MB, SEG), 256, LDS_BYTES, stream>>>(
        qf16, sf16, q2, s2, parts);
    knn_finalize<<<NQ / 256, 256, 0, stream>>>(parts, idx_out, w_out);
}
